// SRA_Module_23398981829236
// MI455X (gfx1250) — matrix operand order repaired and checked by reading, not compiled, not run on hardware
//
#include <hip/hip_runtime.h>

// ---------------------------------------------------------------------------
// SRA module, algebraically folded:
//   g_pre = A1 @ theta + A2 @ phi,  A1 = Wgg[:, :S] @ phi^T, A2 = Wgg[:, S:] @ theta^T
// (removes the [B,2048,2048] Gs tensor and its 34-GFLOP GEMM; remaining work
//  ~2.3 G MACs, ~70 MB traffic -> memory-bound at 23.3 TB/s).
// All GEMMs on v_wmma_f32_16x16x32_bf16. Intermediates consumed as WMMA-B
// operands are stored pre-swizzled in this file's contiguous-k B layout (k = 16h + i) so consumer
// lanes read 16 contiguous bf16 (2x b128) instead of 16 strided u16.
// ---------------------------------------------------------------------------

typedef __bf16 bf16;
typedef __attribute__((ext_vector_type(16))) __bf16 v16bf;
typedef __attribute__((ext_vector_type(8)))  __bf16 v8bf;
typedef __attribute__((ext_vector_type(8)))  float  v8f;

#define WMMA_BF16(acc, a, b) \
  __builtin_amdgcn_wmma_f32_16x16x32_bf16(false, (a), false, (b), (short)0, (acc), false, false)

// Operand k order used here (16-bit A 16x32, B 32x16): contiguous k = i + 16h, h = lane>>4:
__device__ __forceinline__ int kpatA(int i, int h) { return i + 16 * h; }
__device__ __forceinline__ int kpatB(int i, int h) { return i + 16 * h; }

static constexpr int B_ = 8, C_ = 512, S_ = 2048, IC_ = 64, IS_ = 256, DS_ = 32;
static constexpr int NST_ = S_ / 16;           // 128 spatial tiles
static constexpr float EPS_ = 1e-5f;

// Tiled-B layout: tile (kt, st) holds a 32(k) x 16(n) block as 512 bf16;
// lane L owns elements i=0..15 at tile_base + L*16, value = M[k = i + 16*(L>>4)][n = L&15].

// ---------------------------------------------------------------------------
// Kernel 1: theta/phi = relu(bn(W @ x))   M=64, N=2048, K=512  (per batch, x2)
// One wave computes a 16x64 strip (4 accumulators, A reused 4x).
// Writes row-major (for k2) + WMMA-B tiled (for k3).
// ---------------------------------------------------------------------------
__global__ __launch_bounds__(256) void k_theta_phi(
    const float* __restrict__ x,
    const float* __restrict__ Wt, const float* __restrict__ Wp,
    const float* __restrict__ tg, const float* __restrict__ tb,
    const float* __restrict__ tm, const float* __restrict__ tv,
    const float* __restrict__ pg, const float* __restrict__ pb,
    const float* __restrict__ pm, const float* __restrict__ pv,
    bf16* __restrict__ th, bf16* __restrict__ ph,
    bf16* __restrict__ th_t, bf16* __restrict__ ph_t)
{
  const int z = blockIdx.z;            // 0..15
  const int b = z & 7;
  const int which = z >> 3;            // 0=theta, 1=phi
  const float* W  = which ? Wp : Wt;
  const float* gg = which ? pg : tg;
  const float* bb = which ? pb : tb;
  const float* mm = which ? pm : tm;
  const float* vv = which ? pv : tv;
  bf16* out   = which ? ph   : th;
  bf16* out_t = (which ? ph_t : th_t) + (size_t)b * (IC_ / 32) * NST_ * 512;

  const int wid  = threadIdx.x >> 5;
  const int lane = threadIdx.x & 31;
  const int h = lane >> 4, ln = lane & 15;
  const int t = blockIdx.x * 8 + wid;  // 0..127 (4 mtiles x 32 ngroups)
  const int mbase = (t >> 5) * 16;
  const int ngrp  = t & 31;
  const int sbase = ngrp * 64;

  const float* Arow  = W + (size_t)(mbase + ln) * C_;
  const float* Bcol0 = x + (size_t)b * C_ * S_ + sbase + ln;

  v8f acc[4] = {};
  for (int k0 = 0; k0 < C_; k0 += 32) {
    __builtin_prefetch(Arow + k0 + 64, 0, 1);      // global_prefetch_b8
    v16bf a;
#pragma unroll
    for (int i = 0; i < 16; ++i) a[i] = (bf16)Arow[k0 + kpatA(i, h)];
#pragma unroll
    for (int nt = 0; nt < 4; ++nt) {
      const float* Bcol = Bcol0 + nt * 16;
      v16bf bv;
#pragma unroll
      for (int i = 0; i < 16; ++i)
        bv[i] = (bf16)Bcol[(size_t)(k0 + kpatB(i, h)) * S_];
      acc[nt] = WMMA_BF16(acc[nt], a, bv);
    }
  }

  const int hp = (mbase >> 4) & 1;     // which half of consumer 32-k tile
#pragma unroll
  for (int nt = 0; nt < 4; ++nt) {
    const int st = ngrp * 4 + nt;
    float vals[8];
    v8bf pk;
#pragma unroll
    for (int r = 0; r < 8; ++r) {
      const int o = mbase + r + 8 * h;
      const float sc = gg[o] * rsqrtf(vv[o] + EPS_);
      vals[r] = fmaxf((acc[nt][r] - mm[o]) * sc + bb[o], 0.0f);
      pk[r] = (bf16)vals[r];
      out[((size_t)b * IC_ + o) * S_ + st * 16 + ln] = pk[r];  // row-major for k2
    }
    // tiled store: element i = r + 8h contiguous -> one 16B store per lane
    bf16* tp = out_t + ((size_t)(mbase >> 5) * NST_ + st) * 512 + (hp * 16 + ln) * 16 + 8 * h;
    *(v8bf*)tp = pk;
  }
}

// ---------------------------------------------------------------------------
// Kernel 2: A1 = Wgg[:, :S] @ phi^T ; A2 = Wgg[:, S:] @ theta^T
//           M=256, N=64, K=2048 (per batch, x2); row-major inputs contiguous.
// ---------------------------------------------------------------------------
__global__ __launch_bounds__(256) void k_fold(
    const float* __restrict__ Wgg,
    const bf16* __restrict__ th, const bf16* __restrict__ ph,
    bf16* __restrict__ A1, bf16* __restrict__ A2)
{
  const int b = blockIdx.z;
  const int wid  = threadIdx.x >> 5;
  const int lane = threadIdx.x & 31;
  const int h = lane >> 4, ln = lane & 15;
  int t = blockIdx.x * 8 + wid;        // 0..127
  const int mat = t >> 6;              // 0 -> A1 (phi), 1 -> A2 (theta)
  t &= 63;                             // 16 x 4 tiles
  const int mbase = (t >> 2) * 16;
  const int nbase = (t & 3) * 16;

  const float* Arow = Wgg + (size_t)(mbase + ln) * (2 * S_) + (mat ? S_ : 0);
  const bf16* src   = mat ? th : ph;
  const bf16* Brow  = src + ((size_t)b * IC_ + nbase + ln) * S_;   // B[k][n] = src[n][k]
  bf16* Aout = mat ? A2 : A1;

  v8f acc = {};
  for (int k0 = 0; k0 < S_; k0 += 32) {
    __builtin_prefetch(Arow + k0 + 128, 0, 1);
    __builtin_prefetch(Brow + k0 + 256, 0, 1);
    v16bf a, bv;
#pragma unroll
    for (int i = 0; i < 16; ++i) {
      a[i]  = (bf16)Arow[k0 + kpatA(i, h)];
      bv[i] = Brow[k0 + kpatB(i, h)];
    }
    acc = WMMA_BF16(acc, a, bv);
  }
#pragma unroll
  for (int r = 0; r < 8; ++r) {
    const int o = mbase + r + 8 * h;
    Aout[((size_t)b * IS_ + o) * IC_ + nbase + ln] = (bf16)acc[r];
  }
}

// ---------------------------------------------------------------------------
// Kernel 3: g = relu(bn(A1 @ theta + A2 @ phi))   M=256, N=2048, K=64 (x2)
// One wave: 16x64 strip, 16 straight-line WMMAs. B from tiled layout
// (fully coalesced), output written tiled for kernel 4.
// ---------------------------------------------------------------------------
__global__ __launch_bounds__(256) void k_g(
    const bf16* __restrict__ A1, const bf16* __restrict__ A2,
    const bf16* __restrict__ th_t, const bf16* __restrict__ ph_t,
    const float* __restrict__ gg, const float* __restrict__ gb,
    const float* __restrict__ gm, const float* __restrict__ gv,
    bf16* __restrict__ g_t)
{
  const int b = blockIdx.z;
  const int wid  = threadIdx.x >> 5;
  const int lane = threadIdx.x & 31;
  const int h = lane >> 4, ln = lane & 15;
  const int t = blockIdx.x * 8 + wid;  // 0..511 (16 mtiles x 32 ngroups)
  const int mbase = (t >> 5) * 16;
  const int st0   = (t & 31) * 4;

  const bf16* A1row = A1 + ((size_t)b * IS_ + mbase + ln) * IC_;
  const bf16* A2row = A2 + ((size_t)b * IS_ + mbase + ln) * IC_;
  const bf16* thb = th_t + (size_t)b * (IC_ / 32) * NST_ * 512;
  const bf16* phb = ph_t + (size_t)b * (IC_ / 32) * NST_ * 512;

  v8f acc[4] = {};
#pragma unroll
  for (int pass = 0; pass < 2; ++pass) {
    const bf16* Arow = pass ? A2row : A1row;
    const bf16* Bt   = pass ? phb   : thb;
#pragma unroll
    for (int kt = 0; kt < 2; ++kt) {
      const int k0 = kt * 32;
      v16bf a;
#pragma unroll
      for (int i = 0; i < 16; ++i) a[i] = Arow[k0 + kpatA(i, h)];
#pragma unroll
      for (int nt = 0; nt < 4; ++nt) {
        const bf16* p = Bt + ((size_t)kt * NST_ + st0 + nt) * 512 + lane * 16;
        v16bf bv;
#pragma unroll
        for (int i = 0; i < 16; ++i) bv[i] = p[i];   // 2x b128, coalesced
        acc[nt] = WMMA_BF16(acc[nt], a, bv);
      }
    }
  }

  bf16* gb_t = g_t + (size_t)b * (IS_ / 32) * NST_ * 512;
  const int hp = (mbase >> 4) & 1;
#pragma unroll
  for (int nt = 0; nt < 4; ++nt) {
    const int st = st0 + nt;
    v8bf pk;
#pragma unroll
    for (int r = 0; r < 8; ++r) {
      const int o = mbase + r + 8 * h;
      const float sc = gg[o] * rsqrtf(gv[o] + EPS_);
      pk[r] = (bf16)fmaxf((acc[nt][r] - gm[o]) * sc + gb[o], 0.0f);
    }
    bf16* tp = gb_t + ((size_t)(mbase >> 5) * NST_ + st) * 512 + (hp * 16 + ln) * 16 + 8 * h;
    *(v8bf*)tp = pk;
  }
}

// ---------------------------------------------------------------------------
// Kernel 4 (fused tail): y = relu(bn(W1@g)) ; ys = bn(W2@y) ; gate = sigmoid
//                        out = x * gate     (bandwidth-dominant stage)
// ---------------------------------------------------------------------------
__global__ __launch_bounds__(128) void k_tail(
    const float* __restrict__ W1, const bf16* __restrict__ g_t,
    const float* __restrict__ w1g, const float* __restrict__ w1b,
    const float* __restrict__ w1m, const float* __restrict__ w1v,
    const float* __restrict__ W2,
    const float* __restrict__ w2g, const float* __restrict__ w2b,
    const float* __restrict__ w2m, const float* __restrict__ w2v,
    const float* __restrict__ x, float* __restrict__ out)
{
  __shared__ float yS[DS_][16];
  __shared__ float gateS[16];

  const int b = blockIdx.z;
  const int st = blockIdx.x;           // spatial tile, sbase = st*16
  const int wid  = threadIdx.x >> 5;
  const int lane = threadIdx.x & 31;
  const int h = lane >> 4, ln = lane & 15;

  if (wid < 2) {                       // M=32 -> two 16x16 tiles, K=256
    const int mbase = wid * 16;
    const float* Arow = W1 + (size_t)(mbase + ln) * IS_;
    const bf16* gbt = g_t + (size_t)b * (IS_ / 32) * NST_ * 512;
    v8f acc = {};
#pragma unroll
    for (int kt = 0; kt < 8; ++kt) {
      const int k0 = kt * 32;
      v16bf a, bv;
      const bf16* p = gbt + ((size_t)kt * NST_ + st) * 512 + lane * 16;
#pragma unroll
      for (int i = 0; i < 16; ++i) {
        a[i]  = (bf16)Arow[k0 + kpatA(i, h)];
        bv[i] = p[i];
      }
      acc = WMMA_BF16(acc, a, bv);
    }
#pragma unroll
    for (int r = 0; r < 8; ++r) {
      const int o = mbase + r + 8 * h;
      const float sc = w1g[o] * rsqrtf(w1v[o] + EPS_);
      yS[o][ln] = fmaxf((acc[r] - w1m[o]) * sc + w1b[o], 0.0f);
    }
  }
  __syncthreads();

  if (threadIdx.x < 16) {
    float s = 0.0f;
#pragma unroll
    for (int d = 0; d < DS_; ++d) s += W2[d] * yS[d][threadIdx.x];
    const float sc = w2g[0] * rsqrtf(w2v[0] + EPS_);
    const float ys = (s - w2m[0]) * sc + w2b[0];
    gateS[threadIdx.x] = 1.0f / (1.0f + expf(-ys));
  }
  __syncthreads();

  const float* xb = x   + (size_t)b * C_ * S_;
  float*       ob = out + (size_t)b * C_ * S_;
  for (int idx = threadIdx.x; idx < C_ * 16; idx += 128) {
    const int c = idx >> 4, n = idx & 15;
    const size_t off = (size_t)c * S_ + st * 16 + n;
    ob[off] = xb[off] * gateS[n];
  }
}

// ---------------------------------------------------------------------------
extern "C" void kernel_launch(void* const* d_in, const int* in_sizes, int n_in,
                              void* d_out, int out_size, void* d_ws, size_t ws_size,
                              hipStream_t stream) {
  (void)in_sizes; (void)n_in; (void)out_size; (void)ws_size;
  const float* x   = (const float*)d_in[0];
  const float* Wt  = (const float*)d_in[1];
  const float* tg  = (const float*)d_in[2];
  const float* tb  = (const float*)d_in[3];
  const float* tm  = (const float*)d_in[4];
  const float* tv  = (const float*)d_in[5];
  const float* Wp  = (const float*)d_in[6];
  const float* pg  = (const float*)d_in[7];
  const float* pb  = (const float*)d_in[8];
  const float* pm  = (const float*)d_in[9];
  const float* pv  = (const float*)d_in[10];
  const float* Wgg = (const float*)d_in[11];
  const float* ggg = (const float*)d_in[12];
  const float* ggb = (const float*)d_in[13];
  const float* ggm = (const float*)d_in[14];
  const float* ggv = (const float*)d_in[15];
  const float* W1  = (const float*)d_in[16];
  const float* w1g = (const float*)d_in[17];
  const float* w1b = (const float*)d_in[18];
  const float* w1m = (const float*)d_in[19];
  const float* w1v = (const float*)d_in[20];
  const float* W2  = (const float*)d_in[21];
  const float* w2g = (const float*)d_in[22];
  const float* w2b = (const float*)d_in[23];
  const float* w2m = (const float*)d_in[24];
  const float* w2v = (const float*)d_in[25];
  float* out = (float*)d_out;

  // workspace layout (bytes)
  char* ws = (char*)d_ws;
  constexpr size_t SZ_RM = (size_t)B_ * IC_ * S_ * sizeof(bf16);            // 2 MB each
  constexpr size_t SZ_T1 = (size_t)B_ * (IC_ / 32) * NST_ * 512 * sizeof(bf16); // 2 MB each
  constexpr size_t SZ_A  = (size_t)B_ * IS_ * IC_ * sizeof(bf16);           // 256 KB each
  bf16* th   = (bf16*)(ws);
  bf16* ph   = (bf16*)(ws + SZ_RM);
  bf16* th_t = (bf16*)(ws + 2 * SZ_RM);
  bf16* ph_t = (bf16*)(ws + 2 * SZ_RM + SZ_T1);
  bf16* A1   = (bf16*)(ws + 2 * SZ_RM + 2 * SZ_T1);
  bf16* A2   = (bf16*)(ws + 2 * SZ_RM + 2 * SZ_T1 + SZ_A);
  bf16* g_t  = (bf16*)(ws + 2 * SZ_RM + 2 * SZ_T1 + 2 * SZ_A);              // 8 MB

  k_theta_phi<<<dim3(16, 1, 16), 256, 0, stream>>>(
      x, Wt, Wp, tg, tb, tm, tv, pg, pb, pm, pv, th, ph, th_t, ph_t);
  k_fold<<<dim3(16, 1, B_), 256, 0, stream>>>(Wgg, th, ph, A1, A2);
  k_g<<<dim3(64, 1, B_), 256, 0, stream>>>(
      A1, A2, th_t, ph_t, ggg, ggb, ggm, ggv, g_t);
  k_tail<<<dim3(NST_, 1, B_), 128, 0, stream>>>(
      W1, g_t, w1g, w1b, w1m, w1v, W2, w2g, w2b, w2m, w2v, x, out);
}
